// Trans_Encoder_82008105550075
// MI455X (gfx1250) — matrix operand order repaired and checked by reading, not compiled, not run on hardware
//
#include <hip/hip_runtime.h>

typedef _Float16 h16;
typedef __attribute__((ext_vector_type(8)))  _Float16 v8h;
typedef __attribute__((ext_vector_type(16))) _Float16 v16h;
typedef __attribute__((ext_vector_type(8)))  float    v8f;

// Problem constants
#define KF    512            // feature dim (= K = N of every GEMM)
#define NTOK  32768          // B*S = 64*512 rows
#define FF2   262144         // 512*512 elements per weight matrix
#define EPS   1e-5f

static __device__ inline v16h cat16(v8h lo, v8h hi) {
    return __builtin_shufflevector(lo, hi, 0,1,2,3,4,5,6,7,8,9,10,11,12,13,14,15);
}

// ---------------------------------------------------------------------------
// x = emb[text] + pe ; also produce f16 copy for WMMA GEMMs
// ---------------------------------------------------------------------------
__global__ __launch_bounds__(256)
void k_embed(const int* __restrict__ text, const float* __restrict__ emb,
             const float* __restrict__ pe, float* __restrict__ x, h16* __restrict__ xh) {
    int idx = blockIdx.x * 256 + threadIdx.x;      // < NTOK*KF
    int tok = idx >> 9;
    int f   = idx & 511;
    int s   = tok & 511;                           // S == 512
    float v = emb[(size_t)text[tok] * KF + f] + pe[s * KF + f];
    x[idx]  = v;
    xh[idx] = (h16)v;
}

// ---------------------------------------------------------------------------
// Transpose + f16-convert all 24 weight matrices: Wt[mat][n][k] = W[k][n]
// mat = layer*4 + {0:q,1:k,2:v,3:f}
// ---------------------------------------------------------------------------
__global__ __launch_bounds__(256)
void k_prep_w(const float* __restrict__ Wq, const float* __restrict__ Wk,
              const float* __restrict__ Wv, const float* __restrict__ Wf,
              h16* __restrict__ Wt) {
    int idx = blockIdx.x * 256 + threadIdx.x;      // < 24*FF2
    int mat = idx >> 18;
    int rem = idx & (FF2 - 1);
    int k   = rem >> 9;
    int n   = rem & 511;
    int layer = mat >> 2;
    int which = mat & 3;
    const float* src = (which == 0) ? Wq : (which == 1) ? Wk : (which == 2) ? Wv : Wf;
    Wt[(size_t)mat * FF2 + (size_t)n * KF + k] = (h16)src[(size_t)layer * FF2 + (size_t)k * KF + n];
}

// ---------------------------------------------------------------------------
// C[M,512] (f16) = A[M,512] (f16) @ Bt^T + bias   (Bt stored [N,K] row-major)
// Block: 256 thr = 8 waves; block tile 64x256; wave tile 32x64.
// A-panel staged in LDS (padded rows, conflict-free ds_read_b128).
// B fragments software-pipelined one k-step ahead; the loop is FULLY
// unrolled so the double-buffer rotation becomes SSA renaming (no v_mov /
// v_nop overhead) while keeping a full 8-wmma latency gap per B clause.
// ---------------------------------------------------------------------------
__global__ __launch_bounds__(256)
void k_gemm(const h16* __restrict__ A, const h16* __restrict__ Bt,
            const float* __restrict__ bias, h16* __restrict__ C) {
    constexpr int LDA = KF + 8;                    // 520 halves -> 1040B row stride
    __shared__ h16 As[64 * LDA];                   // 66.5 KB

    const int m0 = blockIdx.x * 64;
    const int n0 = blockIdx.y * 256;

    // Cooperative, coalesced load of the 64x512 f16 A panel
    {
        int t = threadIdx.x;
        #pragma unroll
        for (int i = 0; i < 16; ++i) {
            int chunk = t + i * 256;               // 0..4095, 8 halves each
            int r = chunk >> 6;                    // 64 chunks per row
            int c = (chunk & 63) << 3;
            *(v8h*)&As[r * LDA + c] = *(const v8h*)(A + (size_t)(m0 + r) * KF + c);
        }
    }
    __syncthreads();

    const int wave = threadIdx.x >> 5;
    const int lane = threadIdx.x & 31;
    const int mi   = wave >> 2;                    // 0..1 : 32-row M half
    const int ni   = wave & 3;                     // 0..3 : 64-col N strip
    const int kup  = lane >> 4;                    // 0 (lanes 0-15) / 1 (lanes 16-31)
    const int lrow = mi * 32 + (lane & 15);
    const int bcol = n0 + ni * 64 + (lane & 15);

    v8f acc0[4] = {v8f{}, v8f{}, v8f{}, v8f{}};    // rows lrow..+15
    v8f acc1[4] = {v8f{}, v8f{}, v8f{}, v8f{}};    // rows lrow+16..+31

    // B fragment base: lane<16 col, K=+0..7,+16..23 ; lane>=16 col, K=+8..15,+24..31
    const h16* bbase = Bt + (size_t)bcol * KF + kup * 8;

    // Prime the B double buffer for k0 = 0
    v16h bcur[4];
    #pragma unroll
    for (int j = 0; j < 4; ++j) {
        const h16* bj = bbase + j * 16 * KF;
        bcur[j] = cat16(*(const v8h*)bj, *(const v8h*)(bj + 16));
    }

    #pragma unroll
    for (int k0 = 0; k0 < KF - 32; k0 += 32) {
        // A fragments per ISA 16-bit 16x32 layout:
        //  lane<16 : row, K = k0+0..7  (v0-3) and k0+16..23 (v4-7)
        //  lane>=16: row, K = k0+8..15 (v0-3) and k0+24..31 (v4-7)
        const h16* ap0 = &As[lrow * LDA + k0 + kup * 8];
        const h16* ap1 = ap0 + 16 * LDA;
        v16h a0 = cat16(*(const v8h*)ap0, *(const v8h*)(ap0 + 16));
        v16h a1 = cat16(*(const v8h*)ap1, *(const v8h*)(ap1 + 16));

        // Issue next k-step's B loads BEFORE consuming the current fragments
        v16h bnext[4];
        {
            const h16* bp = bbase + (k0 + 32);
            #pragma unroll
            for (int j = 0; j < 4; ++j) {
                const h16* bj = bp + j * 16 * KF;
                bnext[j] = cat16(*(const v8h*)bj, *(const v8h*)(bj + 16));
            }
        }

        #pragma unroll
        for (int j = 0; j < 4; ++j) {
            acc0[j] = __builtin_amdgcn_wmma_f32_16x16x32_f16(
                false, a0, false, bcur[j], (short)0, acc0[j], false, false);
            acc1[j] = __builtin_amdgcn_wmma_f32_16x16x32_f16(
                false, a1, false, bcur[j], (short)0, acc1[j], false, false);
        }
        #pragma unroll
        for (int j = 0; j < 4; ++j) bcur[j] = bnext[j];
    }

    // Final (peeled) k-step: k0 = KF-32, fragments already resident
    {
        const int k0 = KF - 32;
        const h16* ap0 = &As[lrow * LDA + k0 + kup * 8];
        const h16* ap1 = ap0 + 16 * LDA;
        v16h a0 = cat16(*(const v8h*)ap0, *(const v8h*)(ap0 + 16));
        v16h a1 = cat16(*(const v8h*)ap1, *(const v8h*)(ap1 + 16));
        #pragma unroll
        for (int j = 0; j < 4; ++j) {
            acc0[j] = __builtin_amdgcn_wmma_f32_16x16x32_f16(
                false, a0, false, bcur[j], (short)0, acc0[j], false, false);
            acc1[j] = __builtin_amdgcn_wmma_f32_16x16x32_f16(
                false, a1, false, bcur[j], (short)0, acc1[j], false, false);
        }
    }

    // Epilogue: C/D layout -> lane<16: M=i, lane>=16: M=8+i ; N = lane&15
    #pragma unroll
    for (int mm = 0; mm < 2; ++mm) {
        const int rbase = m0 + mi * 32 + mm * 16 + kup * 8;
        #pragma unroll
        for (int j = 0; j < 4; ++j) {
            int cc = n0 + ni * 64 + j * 16 + (lane & 15);
            float bs = bias[cc];
            const v8f& a = mm ? acc1[j] : acc0[j];
            #pragma unroll
            for (int i = 0; i < 8; ++i)
                C[(size_t)(rbase + i) * KF + cc] = (h16)(a[i] + bs);
        }
    }
}

// ---------------------------------------------------------------------------
// Per-token head attention (8x8, no 1/sqrt(d) scaling) + residual + LayerNorm
// One 64-thread block per token (2 waves).
// ---------------------------------------------------------------------------
__global__ __launch_bounds__(64)
void k_attn_ln(const h16* __restrict__ q, const h16* __restrict__ k,
               const h16* __restrict__ v, const float* __restrict__ g1,
               const float* __restrict__ b1, float* __restrict__ x,
               h16* __restrict__ xh) {
    __shared__ float qs[KF], ks[KF], vs[KF];
    __shared__ float sc[64], wsm[64], red[64], red2[64];

    const int tok = blockIdx.x;
    const int t   = threadIdx.x;
    const size_t base = (size_t)tok * KF;

    #pragma unroll
    for (int i = 0; i < 8; ++i) {
        int f = t * 8 + i;
        qs[f] = (float)q[base + f];
        ks[f] = (float)k[base + f];
        vs[f] = (float)v[base + f];
    }
    __syncthreads();

    // score[h][g] = q_h . k_g  over DH=64
    const int h = t >> 3, g = t & 7;
    float s = 0.f;
    #pragma unroll 8
    for (int d = 0; d < 64; ++d) s += qs[h * 64 + d] * ks[g * 64 + d];
    sc[t] = s;
    __syncthreads();

    // softmax over g
    float m = sc[h * 8];
    #pragma unroll
    for (int gg = 1; gg < 8; ++gg) m = fmaxf(m, sc[h * 8 + gg]);
    float denom = 0.f;
    #pragma unroll
    for (int gg = 0; gg < 8; ++gg) denom += __expf(sc[h * 8 + gg] - m);
    wsm[t] = __expf(s - m) / denom;
    __syncthreads();

    // attn + residual; per-thread 8 features
    float r[8];
    float psum = 0.f, psq = 0.f;
    #pragma unroll
    for (int i = 0; i < 8; ++i) {
        int f = t * 8 + i;
        int hh = f >> 6, d = f & 63;
        float a = 0.f;
        #pragma unroll
        for (int gg = 0; gg < 8; ++gg) a += wsm[hh * 8 + gg] * vs[gg * 64 + d];
        float val = a + x[base + f];
        r[i] = val;
        psum += val;
        psq  += val * val;
    }
    red[t] = psum; red2[t] = psq;
    __syncthreads();
    for (int st = 32; st > 0; st >>= 1) {
        if (t < st) { red[t] += red[t + st]; red2[t] += red2[t + st]; }
        __syncthreads();
    }
    float mean = red[0] * (1.f / KF);
    float var  = red2[0] * (1.f / KF) - mean * mean;
    float inv  = rsqrtf(var + EPS);
    #pragma unroll
    for (int i = 0; i < 8; ++i) {
        int f = t * 8 + i;
        float o = (r[i] - mean) * inv * g1[f] + b1[f];
        x[base + f]  = o;
        xh[base + f] = (h16)o;
    }
}

// ---------------------------------------------------------------------------
// x = LN(y + x) with g2/beta2 ; also refresh f16 copy
// ---------------------------------------------------------------------------
__global__ __launch_bounds__(64)
void k_ff_ln(const h16* __restrict__ y, const float* __restrict__ g2,
             const float* __restrict__ b2, float* __restrict__ x,
             h16* __restrict__ xh) {
    __shared__ float red[64], red2[64];
    const int tok = blockIdx.x;
    const int t   = threadIdx.x;
    const size_t base = (size_t)tok * KF;

    float r[8];
    float psum = 0.f, psq = 0.f;
    #pragma unroll
    for (int i = 0; i < 8; ++i) {
        int f = t * 8 + i;
        float val = (float)y[base + f] + x[base + f];
        r[i] = val;
        psum += val;
        psq  += val * val;
    }
    red[t] = psum; red2[t] = psq;
    __syncthreads();
    for (int st = 32; st > 0; st >>= 1) {
        if (t < st) { red[t] += red[t + st]; red2[t] += red2[t + st]; }
        __syncthreads();
    }
    float mean = red[0] * (1.f / KF);
    float var  = red2[0] * (1.f / KF) - mean * mean;
    float inv  = rsqrtf(var + EPS);
    #pragma unroll
    for (int i = 0; i < 8; ++i) {
        int f = t * 8 + i;
        float o = (r[i] - mean) * inv * g2[f] + b2[f];
        x[base + f]  = o;
        xh[base + f] = (h16)o;
    }
}

// ---------------------------------------------------------------------------
extern "C" void kernel_launch(void* const* d_in, const int* in_sizes, int n_in,
                              void* d_out, int out_size, void* d_ws, size_t ws_size,
                              hipStream_t stream) {
    const int*   text = (const int*)  d_in[0];
    const float* emb  = (const float*)d_in[1];
    const float* pe   = (const float*)d_in[2];
    const float* Wq   = (const float*)d_in[3];
    const float* bq   = (const float*)d_in[4];
    const float* Wk   = (const float*)d_in[5];
    const float* bk   = (const float*)d_in[6];
    const float* Wv   = (const float*)d_in[7];
    const float* bv   = (const float*)d_in[8];
    const float* g1   = (const float*)d_in[9];
    const float* be1  = (const float*)d_in[10];
    const float* Wf   = (const float*)d_in[11];
    const float* bf   = (const float*)d_in[12];
    const float* g2   = (const float*)d_in[13];
    const float* be2  = (const float*)d_in[14];

    float* x = (float*)d_out;                      // running activations (fp32)

    const size_t ACT = (size_t)NTOK * KF;          // 16,777,216 elements
    h16* xh = (h16*)d_ws;                          // f16 activations
    h16* qb = xh + ACT;                            // q  (reused as FF output y)
    h16* kb = qb + ACT;
    h16* vb = kb + ACT;
    h16* Wt = vb + ACT;                            // 24 transposed f16 matrices

    // prologue
    k_embed <<<65536, 256, 0, stream>>>(text, emb, pe, x, xh);
    k_prep_w<<<24576, 256, 0, stream>>>(Wq, Wk, Wv, Wf, Wt);

    dim3 ggemm(NTOK / 64, KF / 256, 1);            // (512, 2)
    for (int l = 0; l < 6; ++l) {
        const h16* Wtq = Wt + (size_t)(l * 4 + 0) * FF2;
        const h16* Wtk = Wt + (size_t)(l * 4 + 1) * FF2;
        const h16* Wtv = Wt + (size_t)(l * 4 + 2) * FF2;
        const h16* Wtf = Wt + (size_t)(l * 4 + 3) * FF2;

        k_gemm<<<ggemm, 256, 0, stream>>>(xh, Wtq, bq + l * KF, qb);
        k_gemm<<<ggemm, 256, 0, stream>>>(xh, Wtk, bk + l * KF, kb);
        k_gemm<<<ggemm, 256, 0, stream>>>(xh, Wtv, bv + l * KF, vb);
        k_attn_ln<<<NTOK, 64, 0, stream>>>(qb, kb, vb, g1 + l * KF, be1 + l * KF, x, xh);
        k_gemm<<<ggemm, 256, 0, stream>>>(xh, Wtf, bf + l * KF, qb);      // y -> qb
        k_ff_ln<<<NTOK, 64, 0, stream>>>(qb, g2 + l * KF, be2 + l * KF, x, xh);
    }
}
